// PC_MO_NOT_R_5454608466695
// MI455X (gfx1250) — matrix operand order repaired and checked by reading, not compiled, not run on hardware
//
#include <hip/hip_runtime.h>
#include <stdint.h>

// ---------------------------------------------------------------------------
// MI455X / gfx1250, wave32. WMMA bf16 16x16x32 for all GEMMs, double-buffered
// TDM tile staging with partial s_wait_tensorcnt, LDS-tiled kNN.
// ---------------------------------------------------------------------------

typedef __bf16 bhalf;
typedef bhalf v16bf __attribute__((ext_vector_type(16)));
typedef float v8f   __attribute__((ext_vector_type(8)));
typedef unsigned int uint;
typedef uint v4u  __attribute__((ext_vector_type(4)));
typedef int  v8i  __attribute__((ext_vector_type(8)));
typedef int  v4i  __attribute__((ext_vector_type(4)));

#ifndef __has_builtin
#define __has_builtin(x) 0
#endif

#if defined(__HIP_DEVICE_COMPILE__) && __has_builtin(__builtin_amdgcn_tensor_load_to_lds)
#define HAVE_TDM 1
#else
#define HAVE_TDM 0
#endif

#if defined(__HIP_DEVICE_COMPILE__) && __has_builtin(__builtin_amdgcn_s_wait_tensorcnt)
#define WAIT_TENSORCNT(n) __builtin_amdgcn_s_wait_tensorcnt(n)
#else
#define WAIT_TENSORCNT(n) asm volatile("s_wait_tensorcnt %0" ::"i"(n) : "memory")
#endif

#define TM  64   // block tile M
#define TNB 32   // block tile N (two 16-wide wmma column groups)
#define TK  32   // K step (one bf16 wmma K)

// ---------------------------------------------------------------------------
// Tensor Data Mover: 2D tile Global -> LDS.  D# packing per CDNA5 ISA
// Group0: [1:0]=count=1, [63:32]=lds_addr, [120:64]=global_addr, [127:126]=2
// Group1: [17:16]=data_size(2 => 4B), [79:48]=tensor_dim0, [111:80]=tensor_dim1,
//         [127:112]=tile_dim0, [143:128]=tile_dim1, [207:160]=tensor_dim0_stride
// OOB reads (beyond tensor dims) return zero -> zero-padded tiles.
// ---------------------------------------------------------------------------
#if HAVE_TDM
__device__ __forceinline__ void tdm_load_2d(uint lds_off, const float* gptr,
                                            int rows, int cols, int ld_elems,
                                            int tile_rows, int tile_cols) {
  unsigned long long ga = (unsigned long long)(uintptr_t)gptr;
  v4u g0;
  g0.x = 1u;
  g0.y = lds_off;
  g0.z = (uint)(ga & 0xffffffffull);
  g0.w = (uint)((ga >> 32) & 0x1ffffffull) | (2u << 30);
  uint td0 = (uint)cols, td1 = (uint)rows;
  unsigned long long s0 = (unsigned long long)ld_elems;
  v8i g1;
  g1[0] = (int)(2u << 16);                                    // data_size = 4B
  g1[1] = (int)((td0 & 0xffffu) << 16);
  g1[2] = (int)((td0 >> 16) | ((td1 & 0xffffu) << 16));
  g1[3] = (int)((td1 >> 16) | (((uint)tile_cols & 0xffffu) << 16));
  g1[4] = (int)((uint)tile_rows & 0xffffu);                   // tile_dim1, tile_dim2=0
  g1[5] = (int)(uint)(s0 & 0xffffffffull);
  g1[6] = (int)(uint)((s0 >> 32) & 0xffffull);
  g1[7] = 0;
  v4i z4 = {0, 0, 0, 0};
#if defined(__clang_major__) && __clang_major__ >= 23
  v8i z8 = {0, 0, 0, 0, 0, 0, 0, 0};
  __builtin_amdgcn_tensor_load_to_lds(g0, g1, z4, z4, z8, 0);
#else
  __builtin_amdgcn_tensor_load_to_lds(g0, g1, z4, z4, 0);
#endif
}
#endif

// ---------------------------------------------------------------------------
// WMMA GEMM: C = epilogue(A[MxK] @ B[KxN]).  f32 in memory, bf16 into the
// matrix core, f32 accumulate.  mode bits: 1=+bias  2=*g+beta  4=relu
// 128 threads = 4 wave32.  Block tile 64x32; each wave owns a 16x32 slab =
// two 16x16 accumulators fed by one shared A fragment (2 wmma / wave / tile).
// Double-buffered LDS; TDM loads for tile kt+1 are issued before waiting on
// tile kt with a partial `s_wait_tensorcnt 2` (in-order per-wave TENSORcnt).
// Fragment loads go through volatile LDS pointers so the compiler cannot CSE
// or hoist them (it cannot see the TDM writes to LDS).
// ---------------------------------------------------------------------------
struct __align__(128) GemmTile {
  float A[TM][TK];    // +0     (8192 B)
  float B[TK][TNB];   // +8192  (4096 B)
};                    // sizeof = 12288

__launch_bounds__(128)
__global__ void gemm_wmma_kernel(const float* __restrict__ A, const float* __restrict__ Bm,
                                 const float* __restrict__ bias, const float* __restrict__ gsc,
                                 const float* __restrict__ beta, float* __restrict__ C,
                                 int M, int N, int K, int mode) {
  __shared__ GemmTile tiles[2];
  const int wave = threadIdx.x >> 5;
  const int lane = threadIdx.x & 31;
  const int half = lane >> 4;
  const int l16  = lane & 15;
  const int m0 = blockIdx.y * TM;
  const int n0 = blockIdx.x * TNB;

  v8f acc0 = {0.f, 0.f, 0.f, 0.f, 0.f, 0.f, 0.f, 0.f};
  v8f acc1 = {0.f, 0.f, 0.f, 0.f, 0.f, 0.f, 0.f, 0.f};
  const int nk = (K + TK - 1) / TK;

#if HAVE_TDM
  // one DMA pair per buffer: A tile (64x32) + B tile (32x32)
  auto issue = [&](int buf, int k0) {
    const uint base = (uint)(buf * (int)sizeof(GemmTile));
    int ar = M - m0; if (ar > TM) ar = TM;
    int ac = K - k0; if (ac > TK) ac = TK;
    tdm_load_2d(base, A + (size_t)m0 * K + k0, ar, ac, K, TM, TK);
    int br = K - k0; if (br > TK)  br = TK;
    int bc = N - n0; if (bc > TNB) bc = TNB;
    tdm_load_2d(base + (uint)(TM * TK * 4), Bm + (size_t)k0 * N + n0, br, bc, N, TK, TNB);
  };
  if (threadIdx.x < 32) issue(0, 0);   // prologue: fill buffer 0
#endif

  for (int kt = 0; kt < nk; ++kt) {
    const int cur = kt & 1;
#if HAVE_TDM
    if (threadIdx.x < 32) {            // wave 0 drives the pipeline
      if (kt + 1 < nk) {
        issue(cur ^ 1, (kt + 1) * TK); // overlap: next tile in flight
        WAIT_TENSORCNT(2);             // oldest pair (tile kt) complete
      } else {
        WAIT_TENSORCNT(0);             // drain on last tile
      }
    }
#else
    {
      const int k0 = kt * TK;
      for (int i = threadIdx.x; i < TM * TK; i += 128) {
        int r = i >> 5, c = i & 31;
        int gm = m0 + r, gk = k0 + c;
        tiles[cur].A[r][c] = (gm < M && gk < K) ? A[(size_t)gm * K + gk] : 0.f;
      }
      for (int i = threadIdx.x; i < TK * TNB; i += 128) {
        int r = i >> 5, c = i & 31;
        int gk = k0 + r, gn = n0 + c;
        tiles[cur].B[r][c] = (gk < K && gn < N) ? Bm[(size_t)gk * N + gn] : 0.f;
      }
    }
#endif
    if (kt + 2 < nk)                   // global_prefetch_b8 two tiles ahead
      __builtin_prefetch(A + (size_t)(m0 + 16 * wave) * K + (kt + 2) * TK, 0, 3);
    __syncthreads();                   // tile[cur] visible to all waves

    // Fragments (wave32): A 16x32, B 32x16 x2, both k = 16*half + e.  Volatile so
    // the ds_loads re-execute every iteration (TDM writes are invisible to AA).
    volatile const float* tA = &tiles[cur].A[0][0];
    volatile const float* tB = &tiles[cur].B[0][0];
    v16bf af, bf0, bf1;
    const int mrow = 16 * wave + l16;
#pragma unroll
    for (int e = 0; e < 16; ++e) {
      const int ka = 16 * half + e;
      af[e] = (bhalf)tA[mrow * TK + ka];
      const int kb = 16 * half + e;
      bf0[e] = (bhalf)tB[kb * TNB + l16];
      bf1[e] = (bhalf)tB[kb * TNB + 16 + l16];
    }
    acc0 = __builtin_amdgcn_wmma_f32_16x16x32_bf16(false, af, false, bf0,
                                                   (short)0, acc0, false, false);
    acc1 = __builtin_amdgcn_wmma_f32_16x16x32_bf16(false, af, false, bf1,
                                                   (short)0, acc1, false, false);
    __syncthreads();                   // reads done before tile[cur] is re-filled
  }

  // C/D layout: element r -> local row r + 8*half, col l16
#pragma unroll
  for (int r = 0; r < 8; ++r) {
    const int m = m0 + 16 * wave + r + 8 * half;
    if (m >= M) continue;
#pragma unroll
    for (int h = 0; h < 2; ++h) {
      const int n = n0 + 16 * h + l16;
      if (n < N) {
        float v = h ? acc1[r] : acc0[r];
        if (mode & 1) v += bias[n];
        if (mode & 2) v = v * gsc[n] + beta[n];
        if (mode & 4) v = v > 0.f ? v : 0.f;
        C[(size_t)m * N + n] = v;
      }
    }
  }
}

// ---------------------------------------------------------------------------
// kNN: one thread per query, reference points streamed through LDS tiles,
// register-resident insertion-sorted top-K (all static indexing, no scratch).
// ---------------------------------------------------------------------------
template <int K>
__launch_bounds__(256)
__global__ void knn_kernel(const float* __restrict__ qpos, const float* __restrict__ rpos,
                           int* __restrict__ idx, int nq, int nr) {
  __shared__ float sx[256], sy[256], sz[256];
  const int b = blockIdx.y;
  const int q = blockIdx.x * 256 + threadIdx.x;
  const bool valid = q < nq;
  float qx = 0.f, qy = 0.f, qz = 0.f;
  if (valid) {
    const float* p = qpos + ((size_t)b * nq + q) * 3;
    qx = p[0]; qy = p[1]; qz = p[2];
  }
  float bd[K]; int bi[K];
#pragma unroll
  for (int i = 0; i < K; ++i) { bd[i] = 3.4e38f; bi[i] = 0; }

  for (int base = 0; base < nr; base += 256) {
    const int r = base + threadIdx.x;
    if (r < nr) {
      const float* p = rpos + ((size_t)b * nr + r) * 3;
      sx[threadIdx.x] = p[0]; sy[threadIdx.x] = p[1]; sz[threadIdx.x] = p[2];
    }
    __syncthreads();
    const int lim = (nr - base < 256) ? (nr - base) : 256;
    for (int j = 0; j < lim; ++j) {
      const float dx = qx - sx[j], dy = qy - sy[j], dz = qz - sz[j];
      const float d = dx * dx + dy * dy + dz * dz;
      if (d < bd[K - 1]) {
#pragma unroll
        for (int i = K - 1; i > 0; --i) {
          if (d < bd[i - 1])      { bd[i] = bd[i - 1]; bi[i] = bi[i - 1]; }
          else if (d < bd[i])     { bd[i] = d;         bi[i] = base + j;  }
        }
        if (d < bd[0]) { bd[0] = d; bi[0] = base + j; }
      }
    }
    __syncthreads();
  }
  if (valid) {
    int* o = idx + ((size_t)b * nq + q) * K;
#pragma unroll
    for (int i = 0; i < K; ++i) o[i] = bi[i];
  }
}

// --------------------------- elementwise helpers ---------------------------
__global__ void frame_kernel(const float* __restrict__ in, float* __restrict__ out,
                             int Bb, int Nn, int t) {
  size_t i = (size_t)blockIdx.x * blockDim.x + threadIdx.x;
  size_t total = (size_t)Bb * Nn * 3;
  if (i >= total) return;
  int c = (int)(i % 3); size_t r = i / 3;
  int n = (int)(r % Nn); int b = (int)(r / Nn);
  out[i] = in[(((size_t)t * Bb + b) * 3 + c) * Nn + n];
}

__global__ void subsample_kernel(const float* __restrict__ in, float* __restrict__ out,
                                 int Bb, int nIn, int nOut, int C) {
  size_t i = (size_t)blockIdx.x * blockDim.x + threadIdx.x;
  size_t total = (size_t)Bb * nOut * C;
  if (i >= total) return;
  int c = (int)(i % C); size_t r = i / C;
  int n = (int)(r % nOut); int b = (int)(r / nOut);
  out[i] = in[((size_t)b * nIn + n) * C + c];
}

__global__ void sa_gather_kernel(const float* __restrict__ feat, const float* __restrict__ pos,
                                 const float* __restrict__ ctr, const int* __restrict__ idx,
                                 float* __restrict__ out, int Bb, int nOut, int nIn, int k, int Cf) {
  const int Cout = Cf + 3;
  size_t i = (size_t)blockIdx.x * blockDim.x + threadIdx.x;
  size_t total = (size_t)Bb * nOut * k * Cout;
  if (i >= total) return;
  int c = (int)(i % Cout); size_t r = i / Cout;
  int j = (int)(r % k);    size_t bi = r / k;
  int n = (int)(bi % nOut); int b = (int)(bi / nOut);
  const int id = idx[((size_t)b * nOut + n) * k + j];
  float v;
  if (c < Cf) v = feat[((size_t)b * nIn + id) * Cf + c];
  else {
    const int d = c - Cf;
    v = pos[((size_t)b * nIn + id) * 3 + d] - ctr[((size_t)b * nOut + n) * 3 + d];
  }
  out[i] = v;
}

__global__ void gather_rows_kernel(const float* __restrict__ in, const int* __restrict__ idx,
                                   float* __restrict__ out, int Bb, int nOut, int nIn, int k, int C) {
  size_t i = (size_t)blockIdx.x * blockDim.x + threadIdx.x;
  size_t total = (size_t)Bb * nOut * k * C;
  if (i >= total) return;
  int c = (int)(i % C); size_t r = i / C;
  int j = (int)(r % k); size_t bi = r / k;
  int n = (int)(bi % nOut); int b = (int)(bi / nOut);
  const int id = idx[((size_t)b * nOut + n) * k + j];
  out[i] = in[((size_t)b * nIn + id) * C + c];
}

__global__ void rowmax_kernel(const float* __restrict__ in, float* __restrict__ out,
                              int R, int k, int C, int ldo, int co, int rowStart) {
  size_t i = (size_t)blockIdx.x * blockDim.x + threadIdx.x;
  size_t total = (size_t)R * C;
  if (i >= total) return;
  int c = (int)(i % C); int r = (int)(i / C);
  float m = -3.4e38f;
  for (int j = 0; j < k; ++j) m = fmaxf(m, in[((size_t)r * k + j) * C + c]);
  out[(size_t)(rowStart + r) * ldo + co + c] = m;
}

__global__ void copy_mat_kernel(const float* __restrict__ src, float* __restrict__ dst,
                                int R, int C, int ldsrc, int lddst, int co) {
  size_t i = (size_t)blockIdx.x * blockDim.x + threadIdx.x;
  size_t total = (size_t)R * C;
  if (i >= total) return;
  int c = (int)(i % C); int r = (int)(i / C);
  dst[(size_t)r * lddst + co + c] = src[(size_t)r * ldsrc + c];
}

__global__ void zero_kernel(float* __restrict__ p, size_t n) {
  size_t i = (size_t)blockIdx.x * blockDim.x + threadIdx.x;
  if (i < n) p[i] = 0.f;
}

__global__ void add_kernel(const float* __restrict__ a, const float* __restrict__ b,
                           float* __restrict__ o, size_t n) {
  size_t i = (size_t)blockIdx.x * blockDim.x + threadIdx.x;
  if (i < n) o[i] = a[i] + b[i];
}

// ----------------------------- GAT attention -------------------------------
template <int KT>
__launch_bounds__(256)
__global__ void gat_attn_kernel(const float* __restrict__ hq, const float* __restrict__ hk,
                                const float* __restrict__ av, float* __restrict__ out,
                                int R, int C) {
  const int row  = blockIdx.x * (blockDim.x >> 5) + (threadIdx.x >> 5);
  const int lane = threadIdx.x & 31;
  if (row >= R) return;
  const float* hqr = hq + (size_t)row * C;
  float s = 0.f;
  for (int c = lane; c < C; c += 32) s += hqr[c] * av[c];
#pragma unroll
  for (int o = 16; o; o >>= 1) s += __shfl_xor(s, o, 32);
  const float eq = s;
  float e[KT];
  float m = -3.4e38f;
#pragma unroll
  for (int j = 0; j < KT; ++j) {
    const float* hkj = hk + ((size_t)row * KT + j) * C;
    float t = 0.f;
    for (int c = lane; c < C; c += 32) t += hkj[c] * av[C + c];
#pragma unroll
    for (int o = 16; o; o >>= 1) t += __shfl_xor(t, o, 32);
    float ee = eq + t;
    ee = ee > 0.f ? ee : 0.2f * ee;   // leaky_relu 0.2
    e[j] = ee; m = fmaxf(m, ee);
  }
  float z = 0.f;
#pragma unroll
  for (int j = 0; j < KT; ++j) { e[j] = __expf(e[j] - m); z += e[j]; }
  const float inv = 1.f / z;
  for (int c = lane; c < C; c += 32) {
    float acc = 0.f;
#pragma unroll
    for (int j = 0; j < KT; ++j) acc += (e[j] * inv) * hk[((size_t)row * KT + j) * C + c];
    out[(size_t)row * C + c] = acc;
  }
}

// ------------------------------ LSTM gates ---------------------------------
__global__ void lstm_kernel(float* __restrict__ H, float* __restrict__ Cs,
                            const float* __restrict__ z, int R, int C2) {
  size_t i = (size_t)blockIdx.x * blockDim.x + threadIdx.x;
  size_t total = (size_t)R * C2;
  if (i >= total) return;
  int r = (int)(i / C2), c = (int)(i % C2);
  const float* zr = z + (size_t)r * 4 * C2;
  const float zi = zr[c], zf = zr[C2 + c], zo = zr[2 * C2 + c], zg = zr[3 * C2 + c];
  const float si = 1.f / (1.f + __expf(-zi));
  const float sf = 1.f / (1.f + __expf(-zf));
  const float so = 1.f / (1.f + __expf(-zo));
  const float cn = sf * Cs[i] + si * tanhf(zg);
  Cs[i] = cn;
  H[i]  = so * tanhf(cn);
}

// ------------------- inverse-distance interpolation (fp) -------------------
__launch_bounds__(256)
__global__ void fp_interp_kernel(const float* __restrict__ x, const float* __restrict__ posC,
                                 const float* __restrict__ posF, const int* __restrict__ idx,
                                 float* __restrict__ out, int Bb, int nC, int nF, int k, int C,
                                 int ldo, int co) {
  const int row  = blockIdx.x * (blockDim.x >> 5) + (threadIdx.x >> 5);
  const int lane = threadIdx.x & 31;
  if (row >= Bb * nF) return;
  const int b = row / nF;
  const float* pf = posF + (size_t)row * 3;
  float w = 0.f; int id = 0;
  if (lane < k) {
    id = idx[(size_t)row * k + lane];
    const float* pc = posC + ((size_t)b * nC + id) * 3;
    const float dx = pf[0] - pc[0], dy = pf[1] - pc[1], dz = pf[2] - pc[2];
    w = 1.f / (dx * dx + dy * dy + dz * dz + 1e-8f);
  }
  float tot = w;
#pragma unroll
  for (int o = 16; o; o >>= 1) tot += __shfl_xor(tot, o, 32);
  w /= tot;
  for (int c = lane; c < C; c += 32) {
    float acc = 0.f;
    for (int j = 0; j < k; ++j) {
      const float wj = __shfl(w, j, 32);
      const int   ij = __shfl(id, j, 32);
      acc += wj * x[((size_t)b * nC + ij) * C + c];
    }
    out[(size_t)row * ldo + co + c] = acc;
  }
}

// ----------------------- refine edge-feature gather ------------------------
__global__ void refine_gather_kernel(const float* __restrict__ x, const int* __restrict__ idx,
                                     float* __restrict__ out, int Bb, int Nn, int kmax, int k,
                                     int dil, int Cx, int rowStart, int rowCount) {
  const int C2 = 2 * Cx;
  size_t i = (size_t)blockIdx.x * blockDim.x + threadIdx.x;
  size_t total = (size_t)rowCount * k * C2;
  if (i >= total) return;
  int c = (int)(i % C2); size_t rr = i / C2;
  int j = (int)(rr % k); size_t rl = rr / k;
  const size_t row = (size_t)rowStart + rl;        // global b*N + n
  const int b = (int)(row / Nn);
  const int id = idx[row * kmax + (size_t)j * dil];
  float v;
  if (c < Cx) v = x[row * Cx + c];
  else        v = x[((size_t)b * Nn + id) * Cx + (c - Cx)] - x[row * Cx + (c - Cx)];
  out[i] = v;
}

// ===========================================================================
extern "C" void kernel_launch(void* const* d_in, const int* in_sizes, int n_in,
                              void* d_out, int out_size, void* d_ws, size_t ws_size,
                              hipStream_t stream) {
  (void)in_sizes; (void)n_in; (void)ws_size;
  const int T = 4, Bb = 4, N = 4096;
  const int n1 = 128, n2 = 64, n3 = 32;

  // JAX pytree leaf order (dicts by sorted key):
  // fp10(8) fp21(8) fp32(8) gat1(4:Wb,Wf,ab,af) gat2(4) gat3(4)
  // lstm1(2:W,b) lstm2(2) lstm3(2) refiner(6:Wo,bo,e0W,e0b,e1W,e1b)
  // sa1(12:perlayer W,b,beta,g) sa2(12) sa3(12)
  const int P_FP10 = 1, P_FP21 = 9, P_FP32 = 17;
  const int P_GAT1 = 25, P_GAT2 = 29, P_GAT3 = 33;
  const int P_LSTM1 = 37, P_LSTM2 = 39, P_LSTM3 = 41;
  const int P_REF = 43;
  const int P_SA1 = 49, P_SA2 = 61, P_SA3 = 73;

  auto leaf = [&](int i) -> const float* { return (const float*)d_in[i]; };

  char* wsp = (char*)d_ws;
  size_t off = 0;
  auto fal = [&](size_t e) -> float* { float* p = (float*)(wsp + off); off += ((e * 4 + 255) & ~(size_t)255); return p; };
  auto ial = [&](size_t e) -> int*   { int*   p = (int*)(wsp + off);   off += ((e * 4 + 255) & ~(size_t)255); return p; };
  auto G1  = [&](size_t t) { return dim3((unsigned)((t + 255) / 256)); };

  auto gemm = [&](const float* A, const float* W, const float* b, const float* g,
                  const float* be, float* Co, int M, int Nn, int K, int mode) {
    dim3 grid((Nn + TNB - 1) / TNB, (M + TM - 1) / TM);
    gemm_wmma_kernel<<<grid, 128, 0, stream>>>(A, W, b, g, be, Co, M, Nn, K, mode);
  };
  // layer: relu((x@W + b)*g + beta)  -> mode 7; leaf order W,b,beta,g
  auto runMLP = [&](const float* in, int R, const int* dims, int nl, int base,
                    float* pingA, float* pingB, float* out) {
    const float* cur = in;
    for (int l = 0; l < nl; ++l) {
      float* dst = (l == nl - 1) ? out : ((l & 1) ? pingB : pingA);
      gemm(cur, leaf(base + 4 * l + 0), leaf(base + 4 * l + 1),
           leaf(base + 4 * l + 3), leaf(base + 4 * l + 2), dst,
           R, dims[l + 1], dims[l], 1 | 2 | 4);
      cur = dst;
    }
  };

  // ---------------- encode (last frame) ----------------
  float* xyz = fal((size_t)Bb * N * 3);
  frame_kernel<<<G1((size_t)Bb * N * 3), 256, 0, stream>>>((const float*)d_in[0], xyz, Bb, N, T - 1);
  float* ctr1 = fal((size_t)Bb * n1 * 3);
  subsample_kernel<<<G1((size_t)Bb * n1 * 3), 256, 0, stream>>>(xyz, ctr1, Bb, N, n1, 3);
  float* ctr2 = fal((size_t)Bb * n2 * 3);
  subsample_kernel<<<G1((size_t)Bb * n2 * 3), 256, 0, stream>>>(ctr1, ctr2, Bb, n1, n2, 3);
  float* ctr3 = fal((size_t)Bb * n3 * 3);
  subsample_kernel<<<G1((size_t)Bb * n3 * 3), 256, 0, stream>>>(ctr2, ctr3, Bb, n2, n3, 3);

  float* pingA = fal((size_t)Bb * N * 256);
  float* pingB = fal((size_t)Bb * N * 256);

  // SA1: 4096 -> 128, k=16, ch 6->32->32->64
  int* idx1 = ial((size_t)Bb * n1 * 16);
  knn_kernel<16><<<dim3((n1 + 255) / 256, Bb), 256, 0, stream>>>(ctr1, xyz, idx1, n1, N);
  float* hb1 = fal((size_t)Bb * n1 * 16 * 6);
  sa_gather_kernel<<<G1((size_t)Bb * n1 * 16 * 6), 256, 0, stream>>>(xyz, xyz, ctr1, idx1, hb1, Bb, n1, N, 16, 3);
  { const int d[4] = {6, 32, 32, 64}; runMLP(hb1, Bb * n1 * 16, d, 3, P_SA1, pingA, pingB, pingA); }
  float* a1 = fal((size_t)Bb * n1 * 64);
  rowmax_kernel<<<G1((size_t)Bb * n1 * 64), 256, 0, stream>>>(pingA, a1, Bb * n1, 16, 64, 64, 0, 0);

  // SA2: 128 -> 64, k=8, ch 67->96->96->128
  int* idx2 = ial((size_t)Bb * n2 * 8);
  knn_kernel<8><<<dim3((n2 + 255) / 256, Bb), 256, 0, stream>>>(ctr2, ctr1, idx2, n2, n1);
  float* hb2 = fal((size_t)Bb * n2 * 8 * 67);
  sa_gather_kernel<<<G1((size_t)Bb * n2 * 8 * 67), 256, 0, stream>>>(a1, ctr1, ctr2, idx2, hb2, Bb, n2, n1, 8, 64);
  { const int d[4] = {67, 96, 96, 128}; runMLP(hb2, Bb * n2 * 8, d, 3, P_SA2, pingA, pingB, pingA); }
  float* a2 = fal((size_t)Bb * n2 * 128);
  rowmax_kernel<<<G1((size_t)Bb * n2 * 128), 256, 0, stream>>>(pingA, a2, Bb * n2, 8, 128, 128, 0, 0);

  // SA3: 64 -> 32, k=8, ch 131->128->128->256
  int* idx3 = ial((size_t)Bb * n3 * 8);
  knn_kernel<8><<<dim3((n3 + 255) / 256, Bb), 256, 0, stream>>>(ctr3, ctr2, idx3, n3, n2);
  float* hb3 = fal((size_t)Bb * n3 * 8 * 131);
  sa_gather_kernel<<<G1((size_t)Bb * n3 * 8 * 131), 256, 0, stream>>>(a2, ctr2, ctr3, idx3, hb3, Bb, n3, n2, 8, 128);
  { const int d[4] = {131, 128, 128, 256}; runMLP(hb3, Bb * n3 * 8, d, 3, P_SA3, pingA, pingB, pingA); }
  float* a3 = fal((size_t)Bb * n3 * 256);
  rowmax_kernel<<<G1((size_t)Bb * n3 * 256), 256, 0, stream>>>(pingA, a3, Bb * n3, 8, 256, 256, 0, 0);

  // ---------------- GAT + LSTM per scale ----------------
  float* H1 = fal((size_t)Bb * n1 * 128); float* Cs1 = fal((size_t)Bb * n1 * 128);
  float* H2 = fal((size_t)Bb * n2 * 256); float* Cs2 = fal((size_t)Bb * n2 * 256);
  float* H3 = fal((size_t)Bb * n3 * 512); float* Cs3 = fal((size_t)Bb * n3 * 512);
  zero_kernel<<<G1((size_t)Bb * n1 * 128), 256, 0, stream>>>(H1, (size_t)Bb * n1 * 128);
  zero_kernel<<<G1((size_t)Bb * n1 * 128), 256, 0, stream>>>(Cs1, (size_t)Bb * n1 * 128);
  zero_kernel<<<G1((size_t)Bb * n2 * 256), 256, 0, stream>>>(H2, (size_t)Bb * n2 * 256);
  zero_kernel<<<G1((size_t)Bb * n2 * 256), 256, 0, stream>>>(Cs2, (size_t)Bb * n2 * 256);
  zero_kernel<<<G1((size_t)Bb * n3 * 512), 256, 0, stream>>>(H3, (size_t)Bb * n3 * 512);
  zero_kernel<<<G1((size_t)Bb * n3 * 512), 256, 0, stream>>>(Cs3, (size_t)Bb * n3 * 512);

  auto gatlstm = [&](const float* f, const float* ctr, int n, int kk, int C,
                     int PG, int PL, float* Hst, float* Cst) {
    const int R = Bb * n;
    int* idxg = ial((size_t)R * kk);
    if (kk == 16) knn_kernel<16><<<dim3((n + 255) / 256, Bb), 256, 0, stream>>>(ctr, ctr, idxg, n, n);
    else          knn_kernel<8> <<<dim3((n + 255) / 256, Bb), 256, 0, stream>>>(ctr, ctr, idxg, n, n);
    float* hq  = fal((size_t)R * C);
    float* hkg = fal((size_t)R * kk * C);
    float* fb  = fal((size_t)R * C);
    float* ff  = fal((size_t)R * C);
    // backward: Wb, ab
    gemm(f, leaf(PG + 0), nullptr, nullptr, nullptr, hq, R, C, C, 0);
    gather_rows_kernel<<<G1((size_t)R * kk * C), 256, 0, stream>>>(hq, idxg, hkg, Bb, n, n, kk, C);
    if (kk == 16) gat_attn_kernel<16><<<dim3((R + 7) / 8), 256, 0, stream>>>(hq, hkg, leaf(PG + 2), fb, R, C);
    else          gat_attn_kernel<8> <<<dim3((R + 7) / 8), 256, 0, stream>>>(hq, hkg, leaf(PG + 2), fb, R, C);
    // forward: Wf, af
    gemm(f, leaf(PG + 1), nullptr, nullptr, nullptr, hq, R, C, C, 0);
    gather_rows_kernel<<<G1((size_t)R * kk * C), 256, 0, stream>>>(hq, idxg, hkg, Bb, n, n, kk, C);
    if (kk == 16) gat_attn_kernel<16><<<dim3((R + 7) / 8), 256, 0, stream>>>(hq, hkg, leaf(PG + 3), ff, R, C);
    else          gat_attn_kernel<8> <<<dim3((R + 7) / 8), 256, 0, stream>>>(hq, hkg, leaf(PG + 3), ff, R, C);
    // z = concat(fb, ff, f, H) @ Wl + bl ; gates
    float* zin = fal((size_t)R * 5 * C);
    copy_mat_kernel<<<G1((size_t)R * C), 256, 0, stream>>>(fb, zin, R, C, C, 5 * C, 0);
    copy_mat_kernel<<<G1((size_t)R * C), 256, 0, stream>>>(ff, zin, R, C, C, 5 * C, C);
    copy_mat_kernel<<<G1((size_t)R * C), 256, 0, stream>>>(f,  zin, R, C, C, 5 * C, 2 * C);
    copy_mat_kernel<<<G1((size_t)R * 2 * C), 256, 0, stream>>>(Hst, zin, R, 2 * C, 2 * C, 5 * C, 3 * C);
    float* z = fal((size_t)R * 8 * C);
    gemm(zin, leaf(PL + 0), leaf(PL + 1), nullptr, nullptr, z, R, 8 * C, 5 * C, 1);
    lstm_kernel<<<G1((size_t)R * 2 * C), 256, 0, stream>>>(Hst, Cst, z, R, 2 * C);
  };
  gatlstm(a1, ctr1, n1, 16, 64,  P_GAT1, P_LSTM1, H1, Cs1);
  gatlstm(a2, ctr2, n2, 16, 128, P_GAT2, P_LSTM2, H2, Cs2);
  gatlstm(a3, ctr3, n3, 8,  256, P_GAT3, P_LSTM3, H3, Cs3);

  // ---------------- decode ----------------
  // fp32: 32 -> 64, k=8, concat H2 skip -> 768 -> 512 -> 384
  int* if32 = ial((size_t)Bb * n2 * 8);
  knn_kernel<8><<<dim3((n2 + 255) / 256, Bb), 256, 0, stream>>>(ctr2, ctr3, if32, n2, n3);
  float* h2cat = fal((size_t)Bb * n2 * 768);
  fp_interp_kernel<<<dim3((Bb * n2 + 7) / 8), 256, 0, stream>>>(H3, ctr3, ctr2, if32, h2cat, Bb, n3, n2, 8, 512, 768, 0);
  copy_mat_kernel<<<G1((size_t)Bb * n2 * 256), 256, 0, stream>>>(H2, h2cat, Bb * n2, 256, 256, 768, 512);
  float* h2d = fal((size_t)Bb * n2 * 384);
  { const int d[3] = {768, 512, 384}; runMLP(h2cat, Bb * n2, d, 2, P_FP32, pingA, pingB, h2d); }

  // fp21: 64 -> 128, k=16, concat H1 skip -> 512 -> 384 -> 256
  int* if21 = ial((size_t)Bb * n1 * 16);
  knn_kernel<16><<<dim3((n1 + 255) / 256, Bb), 256, 0, stream>>>(ctr1, ctr2, if21, n1, n2);
  float* h1cat = fal((size_t)Bb * n1 * 512);
  fp_interp_kernel<<<dim3((Bb * n1 + 7) / 8), 256, 0, stream>>>(h2d, ctr2, ctr1, if21, h1cat, Bb, n2, n1, 16, 384, 512, 0);
  copy_mat_kernel<<<G1((size_t)Bb * n1 * 128), 256, 0, stream>>>(H1, h1cat, Bb * n1, 128, 128, 512, 384);
  float* h1d = fal((size_t)Bb * n1 * 256);
  { const int d[3] = {512, 384, 256}; runMLP(h1cat, Bb * n1, d, 2, P_FP21, pingA, pingB, h1d); }

  // fp10: 128 -> 4096, k=32 -> 256 -> 256 -> 128
  int* if10 = ial((size_t)Bb * N * 32);
  knn_kernel<32><<<dim3((N + 255) / 256, Bb), 256, 0, stream>>>(xyz, ctr1, if10, N, n1);
  float* h0cat = fal((size_t)Bb * N * 256);
  fp_interp_kernel<<<dim3((Bb * N + 7) / 8), 256, 0, stream>>>(h1d, ctr1, xyz, if10, h0cat, Bb, n1, N, 32, 256, 256, 0);
  float* h0 = fal((size_t)Bb * N * 128);
  { const int d[3] = {256, 256, 128}; runMLP(h0cat, Bb * N, d, 2, P_FP10, pingA, pingB, h0); }

  // refine: self-kNN kmax=32, dilations {1,2}, edge MLP 256->128, max over 16
  int* idxr = ial((size_t)Bb * N * 32);
  knn_kernel<32><<<dim3((N + 255) / 256, Bb), 256, 0, stream>>>(xyz, xyz, idxr, N, N);
  float* feats = fal((size_t)Bb * N * 256);
  const int CH = 1024;
  float* egin = fal((size_t)CH * 16 * 256);
  float* eout = fal((size_t)CH * 16 * 128);
  for (int di = 0; di < 2; ++di) {
    const int dil = di + 1;
    const float* We = leaf(P_REF + 2 + 2 * di);
    const float* be = leaf(P_REF + 3 + 2 * di);
    for (int rs = 0; rs < Bb * N; rs += CH) {
      refine_gather_kernel<<<G1((size_t)CH * 16 * 256), 256, 0, stream>>>(h0, idxr, egin, Bb, N, 32, 16, dil, 128, rs, CH);
      gemm(egin, We, be, nullptr, nullptr, eout, CH * 16, 128, 256, 1 | 4);
      rowmax_kernel<<<G1((size_t)CH * 128), 256, 0, stream>>>(eout, feats, CH, 16, 128, 256, 128 * di, rs);
    }
  }
  float* delta = fal((size_t)Bb * N * 3);
  gemm(feats, leaf(P_REF + 0), leaf(P_REF + 1), nullptr, nullptr, delta, Bb * N, 3, 256, 1);

  float* outp = (float*)d_out;
  add_kernel<<<G1((size_t)Bb * N * 3), 256, 0, stream>>>(xyz, delta, outp, (size_t)Bb * N * 3);
  const int npred = out_size / (Bb * N * 3);
  for (int p = 1; p < npred; ++p)
    hipMemcpyAsync(outp + (size_t)p * Bb * N * 3, outp, (size_t)Bb * N * 3 * sizeof(float),
                   hipMemcpyDeviceToDevice, stream);
}
